// MoETransformerBlock_81630148428172
// MI455X (gfx1250) — matrix operand order repaired and checked by reading, not compiled, not run on hardware
//
#include <hip/hip_runtime.h>
#include <hip/hip_bf16.h>
#include <math.h>

// ---------------------------------------------------------------------------
// MoE transformer block for gfx1250 (MI455X), bf16 WMMA everywhere.
// B=2 S=512 D=768 H=12 HD=64 E=8 FF=3072, T=B*S=1024 tokens.
// ---------------------------------------------------------------------------

typedef __attribute__((ext_vector_type(16))) __bf16 v16bf;
typedef __attribute__((ext_vector_type(8)))  __bf16 v8bf;
typedef __attribute__((ext_vector_type(8)))  float  v8f;
typedef __attribute__((ext_vector_type(4)))  unsigned int su32x4;
typedef __attribute__((ext_vector_type(8)))  unsigned int su32x8;

#define Tn   1024
#define Dn   768
#define Hn   12
#define HDn  64
#define En   8
#define FFn  3072
#define Sn   512
#define BHn  24
#define TDn  2304   // 3*D

// ---------------------------------------------------------------------------
// Fragment helpers (CDNA5 bf16 WMMA 16x16x32; A/B use this code's contiguous k order, same in both):
//  A (16x32): lanes 0-15 carry row M=lane, K=0..15 contiguous;
//             lanes 16-31 carry K=16..31.
//  B (32x16): lane carries column N=lane&15, K contiguous; lanes 0-15 K=0..15,
//             lanes 16-31 K=16..31.
//  C/D f32:   VGPR r, lanes 0-15 -> M=r, lanes 16-31 -> M=r+8; N=lane&15.
// ---------------------------------------------------------------------------
__device__ __forceinline__ v16bf load_a_frag(const __bf16* ap)
{
    v8bf lo = *(const v8bf*)ap;
    v8bf hi = *(const v8bf*)(ap + 8);
    return __builtin_shufflevector(lo, hi,
             0,1,2,3,4,5,6,7,8,9,10,11,12,13,14,15);
}

// ---------------------------------------------------------------------------
// Direct-from-global WMMA GEMM:
//   C[M,N] = scale * A[M,K] @ B[N,K]^T (+bias[N]) (+res[M,N])
// A,B bf16 row-major (ld=K), C f32/bf16 row-major (ld=N), batched via
// blockIdx.z. 128 threads = 4 waves; block tile 128(M) x 64(N); each wave does
// a 32x64 strip = 2 A-frags x 4 B-frags -> 8 v_wmma per K=32 step.
// Requires M%128==0, N%64==0, K%32==0 (true at every call site).
// ---------------------------------------------------------------------------
template<bool GELU, bool OUT_BF16>
__global__ __launch_bounds__(128)
void wmma_gemm_bt(const __bf16* __restrict__ A, long long sA,
                  const __bf16* __restrict__ B, long long sB,
                  const float*  __restrict__ bias, long long sBias,
                  const float*  __restrict__ resid, long long sRes,
                  void* __restrict__ Cout, long long sC,
                  int M, int N, int K, float scale)
{
    const int batch = blockIdx.z;
    A += (long long)batch * sA;
    B += (long long)batch * sB;
    if (bias)  bias  += (long long)batch * sBias;
    if (resid) resid += (long long)batch * sRes;
    const long long coff = (long long)batch * sC;

    const int lane = threadIdx.x & 31;
    const int wave = threadIdx.x >> 5;
    const int m0 = blockIdx.x * 128 + wave * 32;
    const int n0 = blockIdx.y * 64;

    const int arow = lane & 15;
    const int akb  = (lane >> 4) * 16;
    const int bcol = lane & 15;
    const int bkb  = (lane >> 4) * 16;

    v8f acc[2][4];
#pragma unroll
    for (int i = 0; i < 2; ++i)
#pragma unroll
        for (int j = 0; j < 4; ++j)
#pragma unroll
            for (int r = 0; r < 8; ++r) acc[i][j][r] = 0.0f;

    for (int k0 = 0; k0 < K; k0 += 32) {
        const __bf16* ap0 = A + (long long)(m0 + arow) * K + k0 + akb;
        const __bf16* ap1 = ap0 + 16LL * K;
        v16bf af0 = load_a_frag(ap0);
        v16bf af1 = load_a_frag(ap1);
        __builtin_prefetch(ap0 + 32, 0, 1);   // next K tile -> global_prefetch
        __builtin_prefetch(ap1 + 32, 0, 1);
#pragma unroll
        for (int j = 0; j < 4; ++j) {
            const __bf16* bp = B + (long long)(n0 + j * 16 + bcol) * K + k0 + bkb;
            v16bf bf = *(const v16bf*)bp;     // 32B contiguous, 32B aligned
            acc[0][j] = __builtin_amdgcn_wmma_f32_16x16x32_bf16(
                            false, af0, false, bf, (short)0, acc[0][j], false, false);
            acc[1][j] = __builtin_amdgcn_wmma_f32_16x16x32_bf16(
                            false, af1, false, bf, (short)0, acc[1][j], false, false);
        }
    }

    const int nlane = n0 + (lane & 15);
#pragma unroll
    for (int i = 0; i < 2; ++i) {
        const int mbase = m0 + i * 16 + (lane >> 4) * 8;
#pragma unroll
        for (int j = 0; j < 4; ++j) {
            const int nn = nlane + j * 16;
            const float bv = bias ? bias[nn] : 0.0f;
#pragma unroll
            for (int r = 0; r < 8; ++r) {
                const int mm = mbase + r;
                float v = acc[i][j][r] * scale + bv;
                if (resid) v += resid[(long long)mm * N + nn];
                if (GELU)  v = 0.5f * v * (1.0f + erff(v * 0.70710678118654752f));
                if (OUT_BF16)
                    ((__bf16*)Cout)[coff + (long long)mm * N + nn] = (__bf16)v;
                else
                    ((float*)Cout)[coff + (long long)mm * N + nn] = v;
            }
        }
    }
}

// ---------------------------------------------------------------------------
// TDM descriptor (D#) for a 2D bf16 tile: 64 rows x 32 cols of B[N,K], loaded
// global -> LDS by the Tensor Data Mover (CDNA5 ISA).
// Groups 2/3 zero-filled: legal for <=2D tiles (tile_dim2=0, iterate off).
// ---------------------------------------------------------------------------
__device__ __forceinline__ void tdm_load_b_tile(const __bf16* gptr,
                                                unsigned ldsOff,
                                                unsigned K, unsigned Nrows)
{
    unsigned long long ga = (unsigned long long)(size_t)gptr;
    su32x4 g0;
    g0[0] = 1u;                                   // count=1, user descriptor
    g0[1] = ldsOff;                               // lds_addr (bytes)
    g0[2] = (unsigned)(ga & 0xFFFFFFFFu);         // global_addr[31:0]
    g0[3] = (unsigned)((ga >> 32) & 0x1FFFFFFu)   // global_addr[56:32]
          | 0x80000000u;                          // type=2 ("image")
    su32x8 g1;
    g1[0] = 1u << 16;                             // data_size=1 -> 2 bytes
    g1[1] = (K & 0xFFFFu) << 16;                  // tensor_dim0[15:0]
    g1[2] = ((K >> 16) & 0xFFFFu)                 // tensor_dim0[31:16]
          | ((Nrows & 0xFFFFu) << 16);            // tensor_dim1[15:0]
    g1[3] = ((Nrows >> 16) & 0xFFFFu)             // tensor_dim1[31:16]
          | (32u << 16);                          // tile_dim0 = 32 elements
    g1[4] = 64u;                                  // tile_dim1 = 64, tile_dim2=0
    g1[5] = K;                                    // tensor_dim0_stride[31:0]
    g1[6] = 0u;                                   // stride[47:32], dim1_stride lo
    g1[7] = 0u;                                   // dim1_stride hi
    su32x4 gz;
    gz[0] = 0u; gz[1] = 0u; gz[2] = 0u; gz[3] = 0u;
    asm volatile("tensor_load_to_lds %0, %1, %2, %3"
                 :: "s"(g0), "s"(g1), "s"(gz), "s"(gz)
                 : "memory");
}

// ---------------------------------------------------------------------------
// TDM-staged WMMA GEMM for the MoE (B tile shared by all 4 waves: DMA it once
// per K-step into LDS, waves read fragments with ds_load_b128).
// Same math/layout contract as wmma_gemm_bt (no residual).
// ---------------------------------------------------------------------------
template<bool GELU, bool OUT_BF16>
__global__ __launch_bounds__(128)
void wmma_gemm_bt_tdm(const __bf16* __restrict__ A, long long sA,
                      const __bf16* __restrict__ B, long long sB,
                      const float*  __restrict__ bias, long long sBias,
                      void* __restrict__ Cout, long long sC,
                      int M, int N, int K, float scale)
{
    __shared__ __align__(32) __bf16 btile[64 * 32];   // 4 KB B tile

    const int batch = blockIdx.z;
    A += (long long)batch * sA;
    B += (long long)batch * sB;
    if (bias) bias += (long long)batch * sBias;
    const long long coff = (long long)batch * sC;

    const int lane = threadIdx.x & 31;
    const int wave = threadIdx.x >> 5;
    const int m0 = blockIdx.x * 128 + wave * 32;
    const int n0 = blockIdx.y * 64;

    const int arow = lane & 15;
    const int akb  = (lane >> 4) * 16;
    const int bcol = lane & 15;
    const int bkb  = (lane >> 4) * 16;
    const unsigned ldsOff = (unsigned)(size_t)(&btile[0]);

    v8f acc[2][4];
#pragma unroll
    for (int i = 0; i < 2; ++i)
#pragma unroll
        for (int j = 0; j < 4; ++j)
#pragma unroll
            for (int r = 0; r < 8; ++r) acc[i][j][r] = 0.0f;

    for (int k0 = 0; k0 < K; k0 += 32) {
        __syncthreads();                       // tile consumed by all waves
        if (threadIdx.x == 0) {                // wave 0 drives the TDM
            tdm_load_b_tile(B + (long long)n0 * K + k0, ldsOff,
                            (unsigned)K, (unsigned)N);
            __builtin_amdgcn_s_wait_tensorcnt(0);
        }
        __syncthreads();                       // tile visible to all waves

        const __bf16* ap0 = A + (long long)(m0 + arow) * K + k0 + akb;
        const __bf16* ap1 = ap0 + 16LL * K;
        v16bf af0 = load_a_frag(ap0);
        v16bf af1 = load_a_frag(ap1);
        __builtin_prefetch(ap0 + 32, 0, 1);
        __builtin_prefetch(ap1 + 32, 0, 1);
#pragma unroll
        for (int j = 0; j < 4; ++j) {
            // LDS tile is row-major 64x32: row r at byte offset r*64.
            v16bf bf = *(const v16bf*)(btile + (j * 16 + bcol) * 32 + bkb);
            acc[0][j] = __builtin_amdgcn_wmma_f32_16x16x32_bf16(
                            false, af0, false, bf, (short)0, acc[0][j], false, false);
            acc[1][j] = __builtin_amdgcn_wmma_f32_16x16x32_bf16(
                            false, af1, false, bf, (short)0, acc[1][j], false, false);
        }
    }

    const int nlane = n0 + (lane & 15);
#pragma unroll
    for (int i = 0; i < 2; ++i) {
        const int mbase = m0 + i * 16 + (lane >> 4) * 8;
#pragma unroll
        for (int j = 0; j < 4; ++j) {
            const int nn = nlane + j * 16;
            const float bv = bias ? bias[nn] : 0.0f;
#pragma unroll
            for (int r = 0; r < 8; ++r) {
                const int mm = mbase + r;
                float v = acc[i][j][r] * scale + bv;
                if (GELU) v = 0.5f * v * (1.0f + erff(v * 0.70710678118654752f));
                if (OUT_BF16)
                    ((__bf16*)Cout)[coff + (long long)mm * N + nn] = (__bf16)v;
                else
                    ((float*)Cout)[coff + (long long)mm * N + nn] = v;
            }
        }
    }
}

// ---------------------------------------------------------------------------
// LayerNorm over D=768, one block (256 threads) per token, bf16 output.
// ---------------------------------------------------------------------------
__global__ __launch_bounds__(256)
void layernorm_bf16(const float* __restrict__ x, const float* __restrict__ g,
                    const float* __restrict__ b, __bf16* __restrict__ out)
{
    const int t   = blockIdx.x;
    const int tid = threadIdx.x;
    const float* xr = x + (long long)t * Dn;
    float a0 = xr[tid], a1 = xr[tid + 256], a2 = xr[tid + 512];
    __shared__ float s1[256], s2[256];
    s1[tid] = a0 + a1 + a2;
    s2[tid] = a0 * a0 + a1 * a1 + a2 * a2;
    __syncthreads();
    for (int st = 128; st > 0; st >>= 1) {
        if (tid < st) { s1[tid] += s1[tid + st]; s2[tid] += s2[tid + st]; }
        __syncthreads();
    }
    const float mu   = s1[0] * (1.0f / Dn);
    const float var  = s2[0] * (1.0f / Dn) - mu * mu;
    const float rstd = rsqrtf(var + 1e-5f);
    __bf16* o = out + (long long)t * Dn;
    o[tid]       = (__bf16)((a0 - mu) * rstd * g[tid]       + b[tid]);
    o[tid + 256] = (__bf16)((a1 - mu) * rstd * g[tid + 256] + b[tid + 256]);
    o[tid + 512] = (__bf16)((a2 - mu) * rstd * g[tid + 512] + b[tid + 512]);
}

// f32 -> bf16, same layout
__global__ void convert_bf16(const float* __restrict__ in,
                             __bf16* __restrict__ out, long long n)
{
    long long i = (long long)blockIdx.x * 256 + threadIdx.x;
    if (i < n) out[i] = (__bf16)in[i];
}

// f32 [Bn,R,C] -> bf16 [Bn,C,R] (per-expert weight transpose for B[N,K] GEMM)
__global__ void transpose_bf16(const float* __restrict__ in,
                               __bf16* __restrict__ out, int R, int C,
                               long long total)
{
    long long i = (long long)blockIdx.x * 256 + threadIdx.x;
    if (i >= total) return;
    const long long rc = (long long)R * C;
    const long long e = i / rc, rem = i % rc;
    const int r = (int)(rem / C), c = (int)(rem % C);
    out[e * rc + (long long)c * R + r] = (__bf16)in[i];
}

// qkv f32 [T, 3D] -> q,k bf16 [BH, S, HD] and v^T bf16 [BH, HD, S]
__global__ __launch_bounds__(768)
void split_qkv(const float* __restrict__ qkv, __bf16* __restrict__ q,
               __bf16* __restrict__ k, __bf16* __restrict__ vt)
{
    const int t = blockIdx.x, c = threadIdx.x;
    const int b = t >> 9, s = t & 511, h = c >> 6, d = c & 63;
    const long long bh = (long long)b * Hn + h;
    const long long src = (long long)t * TDn + c;
    q [(bh * Sn + s) * HDn + d] = (__bf16)qkv[src];
    k [(bh * Sn + s) * HDn + d] = (__bf16)qkv[src + Dn];
    vt[(bh * HDn + d) * Sn + s] = (__bf16)qkv[src + 2 * Dn];
}

// row softmax over 512 cols, bf16 probabilities out. One block per row.
__global__ __launch_bounds__(256)
void softmax_rows(const float* __restrict__ sc, __bf16* __restrict__ p)
{
    const long long r = blockIdx.x;
    const int tid = threadIdx.x;
    const float* row = sc + r * Sn;
    float v0 = row[tid], v1 = row[tid + 256];
    __shared__ float red[256];
    red[tid] = fmaxf(v0, v1);
    __syncthreads();
    for (int st = 128; st > 0; st >>= 1) {
        if (tid < st) red[tid] = fmaxf(red[tid], red[tid + st]);
        __syncthreads();
    }
    const float mx = red[0];
    __syncthreads();
    const float e0 = __expf(v0 - mx), e1 = __expf(v1 - mx);
    red[tid] = e0 + e1;
    __syncthreads();
    for (int st = 128; st > 0; st >>= 1) {
        if (tid < st) red[tid] += red[tid + st];
        __syncthreads();
    }
    const float inv = 1.0f / red[0];
    p[r * Sn + tid]       = (__bf16)(e0 * inv);
    p[r * Sn + tid + 256] = (__bf16)(e1 * inv);
}

// attn f32 [BH, S, HD] -> merged bf16 [T, D]
__global__ __launch_bounds__(768)
void merge_heads(const float* __restrict__ at, __bf16* __restrict__ o)
{
    const int t = blockIdx.x, c = threadIdx.x;
    const int b = t >> 9, s = t & 511, h = c >> 6, d = c & 63;
    o[(long long)t * Dn + c] =
        (__bf16)at[(((long long)b * Hn + h) * Sn + s) * HDn + d];
}

// top-2 router with renormalized gates -> comb [T, E]
__global__ __launch_bounds__(256)
void router_top2(const __bf16* __restrict__ h2, const float* __restrict__ rw,
                 const float* __restrict__ rb, float* __restrict__ comb)
{
    const int t = blockIdx.x;
    const int lane = threadIdx.x & 31;
    const int e = threadIdx.x >> 5;                 // 8 waves, 1 expert each
    float s = 0.0f;
    for (int d = lane; d < Dn; d += 32)
        s += (float)h2[(long long)t * Dn + d] * rw[e * Dn + d];
    for (int m = 16; m > 0; m >>= 1) s += __shfl_xor(s, m, 32);
    __shared__ float logits[En];
    if (lane == 0) logits[e] = s + rb[e];
    __syncthreads();
    if (threadIdx.x == 0) {
        float mx = logits[0];
        for (int i = 1; i < En; ++i) mx = fmaxf(mx, logits[i]);
        float p[En];
        for (int i = 0; i < En; ++i) p[i] = __expf(logits[i] - mx);
        int i1 = 0;
        for (int i = 1; i < En; ++i) if (p[i] > p[i1]) i1 = i;
        int i2 = (i1 == 0) ? 1 : 0;
        for (int i = 0; i < En; ++i) if (i != i1 && p[i] > p[i2]) i2 = i;
        const float w = p[i1] + p[i2];
        for (int i = 0; i < En; ++i)
            comb[(long long)t * En + i] =
                (i == i1) ? p[i1] / w : ((i == i2) ? p[i2] / w : 0.0f);
    }
}

// out = x1 + sum_e comb[t,e] * y[e,t,:]
__global__ void moe_combine(const float* __restrict__ x1,
                            const float* __restrict__ y,
                            const float* __restrict__ comb,
                            float* __restrict__ out)
{
    long long i = (long long)blockIdx.x * 256 + threadIdx.x;
    if (i >= (long long)Tn * Dn) return;
    const long long t = i / Dn;
    const int c = (int)(i % Dn);
    float acc = x1[i];
#pragma unroll
    for (int e = 0; e < En; ++e)
        acc += comb[t * En + e] * y[((long long)e * Tn + t) * Dn + c];
    out[i] = acc;
}

// ---------------------------------------------------------------------------
extern "C" void kernel_launch(void* const* d_in, const int* in_sizes, int n_in,
                              void* d_out, int out_size, void* d_ws,
                              size_t ws_size, hipStream_t stream)
{
    const float* x      = (const float*)d_in[0];
    const float* ln1_g  = (const float*)d_in[1];
    const float* ln1_b  = (const float*)d_in[2];
    const float* ln2_g  = (const float*)d_in[3];
    const float* ln2_b  = (const float*)d_in[4];
    const float* wqkv   = (const float*)d_in[5];   // [3D, D]
    const float* bqkv   = (const float*)d_in[6];
    const float* wo     = (const float*)d_in[7];   // [D, D]
    const float* bo     = (const float*)d_in[8];
    const float* rw     = (const float*)d_in[9];   // [E, D]
    const float* rb     = (const float*)d_in[10];
    const float* w1     = (const float*)d_in[11];  // [E, D, FF]
    const float* b1     = (const float*)d_in[12];  // [E, FF]
    const float* w2     = (const float*)d_in[13];  // [E, FF, D]
    const float* b2     = (const float*)d_in[14];  // [E, D]
    float* out = (float*)d_out;

    char* ws = (char*)d_ws;
    size_t off = 0;
    auto alloc = [&](size_t bytes) -> void* {
        void* p = ws + off;
        off += (bytes + 255) & ~(size_t)255;
        return p;
    };
    __bf16* h1b   = (__bf16*)alloc((size_t)Tn * Dn * 2);
    __bf16* wqkvb = (__bf16*)alloc((size_t)TDn * Dn * 2);
    __bf16* wob   = (__bf16*)alloc((size_t)Dn * Dn * 2);
    __bf16* w1t   = (__bf16*)alloc((size_t)En * FFn * Dn * 2);  // [E,FF,D]
    __bf16* w2t   = (__bf16*)alloc((size_t)En * Dn * FFn * 2);  // [E,D,FF]
    float*  qkvf  = (float*) alloc((size_t)Tn * TDn * 4);
    __bf16* qb    = (__bf16*)alloc((size_t)BHn * Sn * HDn * 2);
    __bf16* kb    = (__bf16*)alloc((size_t)BHn * Sn * HDn * 2);
    __bf16* vtb   = (__bf16*)alloc((size_t)BHn * HDn * Sn * 2);
    float*  scf   = (float*) alloc((size_t)BHn * Sn * Sn * 4);
    __bf16* pbb   = (__bf16*)alloc((size_t)BHn * Sn * Sn * 2);
    float*  atf   = (float*) alloc((size_t)BHn * Sn * HDn * 4);
    __bf16* ob    = (__bf16*)alloc((size_t)Tn * Dn * 2);
    float*  x1f   = (float*) alloc((size_t)Tn * Dn * 4);
    __bf16* h2b   = (__bf16*)alloc((size_t)Tn * Dn * 2);
    float*  combf = (float*) alloc((size_t)Tn * En * 4);
    __bf16* hidb  = (__bf16*)alloc((size_t)En * Tn * FFn * 2);
    float*  yf    = (float*) alloc((size_t)En * Tn * Dn * 4);
    (void)ws_size; (void)in_sizes; (void)n_in; (void)out_size;

    // ---- attention sublayer -------------------------------------------------
    layernorm_bf16<<<Tn, 256, 0, stream>>>(x, ln1_g, ln1_b, h1b);

    { long long n = (long long)TDn * Dn;
      convert_bf16<<<(int)((n + 255) / 256), 256, 0, stream>>>(wqkv, wqkvb, n); }
    { long long n = (long long)Dn * Dn;
      convert_bf16<<<(int)((n + 255) / 256), 256, 0, stream>>>(wo, wob, n); }
    { long long n = (long long)En * Dn * FFn;
      transpose_bf16<<<(int)((n + 255) / 256), 256, 0, stream>>>(w1, w1t, Dn, FFn, n);
      transpose_bf16<<<(int)((n + 255) / 256), 256, 0, stream>>>(w2, w2t, FFn, Dn, n); }

    // qkv = h1 @ Wqkv^T + b
    wmma_gemm_bt<false, false><<<dim3(Tn / 128, TDn / 64, 1), 128, 0, stream>>>(
        h1b, 0, wqkvb, 0, bqkv, 0, nullptr, 0, qkvf, 0, Tn, TDn, Dn, 1.0f);

    split_qkv<<<Tn, 768, 0, stream>>>(qkvf, qb, kb, vtb);

    // scores = (q @ k^T) / sqrt(64), batched over 24 (b,h)
    wmma_gemm_bt<false, false><<<dim3(Sn / 128, Sn / 64, BHn), 128, 0, stream>>>(
        qb, (long long)Sn * HDn, kb, (long long)Sn * HDn, nullptr, 0, nullptr, 0,
        scf, (long long)Sn * Sn, Sn, Sn, HDn, 0.125f);

    softmax_rows<<<BHn * Sn, 256, 0, stream>>>(scf, pbb);

    // attn = P @ v  (B = v^T stored [HD, S])
    wmma_gemm_bt<false, false><<<dim3(Sn / 128, HDn / 64, BHn), 128, 0, stream>>>(
        pbb, (long long)Sn * Sn, vtb, (long long)HDn * Sn, nullptr, 0, nullptr, 0,
        atf, (long long)Sn * HDn, Sn, HDn, Sn, 1.0f);

    merge_heads<<<Tn, 768, 0, stream>>>(atf, ob);

    // x1 = x + o @ Wo^T + bo   (residual fused into epilogue)
    wmma_gemm_bt<false, false><<<dim3(Tn / 128, Dn / 64, 1), 128, 0, stream>>>(
        ob, 0, wob, 0, bo, 0, x, 0, x1f, 0, Tn, Dn, Dn, 1.0f);

    // ---- MoE sublayer -------------------------------------------------------
    layernorm_bf16<<<Tn, 256, 0, stream>>>(x1f, ln2_g, ln2_b, h2b);
    router_top2<<<Tn, 256, 0, stream>>>(h2b, rw, rb, combf);

    // hid[e] = gelu(h2 @ w1[e] + b1[e]) -> bf16 ; B tiles staged by the TDM
    wmma_gemm_bt_tdm<true, true><<<dim3(Tn / 128, FFn / 64, En), 128, 0, stream>>>(
        h2b, 0, w1t, (long long)FFn * Dn, b1, (long long)FFn,
        hidb, (long long)Tn * FFn, Tn, FFn, Dn, 1.0f);

    // y[e] = hid[e] @ w2[e] + b2[e]
    wmma_gemm_bt_tdm<false, false><<<dim3(Tn / 128, Dn / 64, En), 128, 0, stream>>>(
        hidb, (long long)Tn * FFn, w2t, (long long)Dn * FFn, b2, (long long)Dn,
        yf, (long long)Tn * Dn, Tn, Dn, FFn, 1.0f);

    moe_combine<<<(Tn * Dn + 255) / 256, 256, 0, stream>>>(x1f, yf, combf, out);
}
